// ENLA_6932077215783
// MI455X (gfx1250) — compile-verified, matrix operand order checked by reading, not run on hardware
//
#include <hip/hip_runtime.h>
#include <hip/hip_bf16.h>
#include <math.h>

// ---------------------------------------------------------------- types
typedef __bf16 bf16;
typedef __attribute__((ext_vector_type(16))) __bf16 v16bf;
typedef __attribute__((ext_vector_type(8)))  float  v8f;

// ---------------------------------------------------------------- shapes
#define B_      4
#define H_      16
#define BH      (B_ * H_)
#define NSEQ    4096
#define DDIM    64
#define MFEAT   266
#define MPAD    288            // 18 tiles of 16 == 9 WMMA K-steps of 32
#define NMT     (MPAD / 16)    // 18
#define CHUNK   128            // n rows per block
#define NCHUNKS (NSEQ / CHUNK) // 32
#define KEPS    1e-4f
#define RATIO   0.06131393f    // 266^-0.5

// LDS pitches (bf16 units; all multiples of 8 -> 16B-aligned rows)
#define XPITCH  64             // staged K/Q tiles [16][64]
#define KTPITCH 136            // kpT / vT, n-direction (128 + 8 pad)
#define QPITCH  296            // qp / ctxT, m-direction (288 + 8 pad)

// ---------------------------------------------------------------- helpers
__device__ __forceinline__ void wave_lds_fence() {
    __builtin_amdgcn_wave_barrier();
    asm volatile("s_wait_dscnt 0" ::: "memory");
    __builtin_amdgcn_wave_barrier();
}

// Per-lane fragment load for bf16 WMMA.
// Works for A from [M][K] row-major and for B from [N][K] (i.e. columns of B
// contiguous): lane L reads row (L&15), two b128 loads at K = h8 and K = 16+h8.
__device__ __forceinline__ v16bf load_tile(const bf16* base, int pitch,
                                           int kbase, int lane) {
    const int row = lane & 15;
    const int h8  = (lane >> 4) << 3;
    const bf16* p = base + row * pitch + kbase + h8;
    union { v16bf v; uint4 q[2]; } u;
    u.q[0] = *(const uint4*)(p);
    u.q[1] = *(const uint4*)(p + 16);
    return u.v;
}

__device__ __forceinline__ v8f wmma_bf16(v16bf a, v16bf b, v8f c) {
    return __builtin_amdgcn_wmma_f32_16x16x32_bf16(false, a, false, b,
                                                   (short)0, c, false, false);
}

// Load one 16x64 f32 tile row-wise, convert to bf16 into LDS stage, and
// produce diag[r] = 0.5*sum_d x[r,d]^2 in diag_s[wave*16 + r].
__device__ __forceinline__ void stage_tile_f32(const float* __restrict__ gsrc,
                                               bf16* xs, float* diag_s,
                                               int wave, int lane) {
    const int row  = lane & 15;
    const int half = lane >> 4;
    const float* rp = gsrc + (size_t)row * DDIM + half * 32;
    float ssq = 0.f;
#pragma unroll
    for (int i = 0; i < 8; ++i) {
        float4 f = ((const float4*)rp)[i];
        ssq += f.x * f.x + f.y * f.y + f.z * f.z + f.w * f.w;
        union { bf16 h[4]; uint2 u; } t;
        t.h[0] = (bf16)f.x; t.h[1] = (bf16)f.y;
        t.h[2] = (bf16)f.z; t.h[3] = (bf16)f.w;
        *(uint2*)(xs + row * XPITCH + half * 32 + i * 4) = t.u;
    }
    ssq += __shfl_xor(ssq, 16);
    if (half == 0) diag_s[wave * 16 + row] = 0.5f * ssq;
}

// ---------------------------------------------------------------- kernel 0
__global__ __launch_bounds__(256) void perf_prep(
    const float* __restrict__ projf, bf16* __restrict__ projbf,
    float* __restrict__ ctx, float* __restrict__ kcum) {
    const int idx = blockIdx.x * 256 + threadIdx.x;
    if (idx < BH * MPAD * DDIM) ctx[idx] = 0.f;
    if (idx < BH * MPAD)        kcum[idx] = 0.f;
    if (idx < MPAD * DDIM) {
        const int m = idx >> 6, d = idx & 63;
        projbf[idx] = (bf16)((m < MFEAT) ? projf[m * DDIM + d] : 0.f);
    }
}

// ---------------------------------------------------------------- kernel 1
// k' features, k_cumsum, ctx = k'^T @ V   (partial per n-chunk, atomics)
__global__ __launch_bounds__(256) void perf_kside(
    const float* __restrict__ kg, const float* __restrict__ vg,
    const bf16* __restrict__ projbf,
    float* __restrict__ ctx, float* __restrict__ kcum) {
    extern __shared__ char smem[];
    bf16*  proj_s = (bf16*)smem;                    // [MPAD][64]
    bf16*  vT     = proj_s + MPAD * XPITCH;         // [64][KTPITCH]
    bf16*  kpT    = vT + 64 * KTPITCH;              // [MPAD][KTPITCH]
    bf16*  xstg   = kpT + MPAD * KTPITCH;           // [8][16][64]
    float* diag_s = (float*)(xstg + 8 * 16 * XPITCH);
    float* kcum_s = diag_s + 8 * 16;                // [MPAD]

    const int tid  = threadIdx.x;
    const int wave = tid >> 5;
    const int lane = tid & 31;
    const int row  = lane & 15;
    const int half = lane >> 4;
    const int bh    = blockIdx.x >> 5;
    const int chunk = blockIdx.x & (NCHUNKS - 1);

    { // proj -> LDS, zero kcum accumulator
        const uint4* src = (const uint4*)projbf;
        uint4* dst = (uint4*)proj_s;
        for (int i = tid; i < MPAD * XPITCH / 8; i += 256) dst[i] = src[i];
        for (int i = tid; i < MPAD; i += 256) kcum_s[i] = 0.f;
    }
    __syncthreads();

    const int    n0   = chunk * CHUNK + wave * 16;
    const size_t base = (size_t)bh * NSEQ * DDIM + (size_t)n0 * DDIM;
    bf16* xs = xstg + wave * 16 * XPITCH;

    stage_tile_f32(kg + base, xs, diag_s, wave, lane);

    { // V tile: transpose-convert into vT[e][n_local]
        const float* rp = vg + base + (size_t)row * DDIM + half * 32;
        const int ncol = wave * 16 + row;
#pragma unroll
        for (int i = 0; i < 8; ++i) {
            float4 f = ((const float4*)rp)[i];
            const int e = half * 32 + i * 4;
            vT[(e + 0) * KTPITCH + ncol] = (bf16)f.x;
            vT[(e + 1) * KTPITCH + ncol] = (bf16)f.y;
            vT[(e + 2) * KTPITCH + ncol] = (bf16)f.z;
            vT[(e + 3) * KTPITCH + ncol] = (bf16)f.w;
        }
    }
    wave_lds_fence();

    float dj[8];
#pragma unroll
    for (int j = 0; j < 8; ++j) dj[j] = diag_s[wave * 16 + j + half * 8];

    const v16bf a0 = load_tile(xs, XPITCH, 0,  lane);
    const v16bf a1 = load_tile(xs, XPITCH, 32, lane);

    for (int mt = 0; mt < NMT; ++mt) {
        v8f acc = {0.f, 0.f, 0.f, 0.f, 0.f, 0.f, 0.f, 0.f};
        v16bf b0 = load_tile(proj_s + mt * 16 * XPITCH, XPITCH, 0,  lane);
        v16bf b1 = load_tile(proj_s + mt * 16 * XPITCH, XPITCH, 32, lane);
        acc = wmma_bf16(a0, b0, acc);
        acc = wmma_bf16(a1, b1, acc);
        const int m = mt * 16 + row;                 // D col -> feature index
        // branchless masking: padded features contribute exactly zero
        const float mRatio = (m < MFEAT) ? RATIO : 0.f;
        float csum = 0.f;
        union { bf16 h[8]; uint4 q; } pk;            // 8 contiguous n-values
#pragma unroll
        for (int j = 0; j < 8; ++j) {                // D row -> n = j + 8*half
            const float val = mRatio * (__expf(acc[j] - dj[j]) + KEPS);
            csum += val;
            pk.h[j] = (bf16)val;
        }
        // transposed store: one aligned b128 per lane
        *(uint4*)(kpT + m * KTPITCH + wave * 16 + half * 8) = pk.q;
        csum += __shfl_xor(csum, 16);
        if (half == 0) atomicAdd(&kcum_s[m], csum);  // ds_add_f32
    }
    __syncthreads();

    for (int i = tid; i < MPAD; i += 256)
        unsafeAtomicAdd(&kcum[bh * MPAD + i], kcum_s[i]);

    // ctx[m,e] += k'^T @ V over this 128-row chunk
    float* ctxp = ctx + (size_t)bh * MPAD * DDIM;
    for (int t = wave; t < NMT * 4; t += 8) {
        const int mt = t >> 2, et = t & 3;
        v8f acc = {0.f, 0.f, 0.f, 0.f, 0.f, 0.f, 0.f, 0.f};
#pragma unroll
        for (int ks = 0; ks < 4; ++ks) {
            v16bf a = load_tile(kpT + mt * 16 * KTPITCH, KTPITCH, ks * 32, lane);
            v16bf b = load_tile(vT  + et * 16 * KTPITCH, KTPITCH, ks * 32, lane);
            acc = wmma_bf16(a, b, acc);
        }
        const int mrow = mt * 16 + half * 8;
        const int e    = et * 16 + row;
#pragma unroll
        for (int j = 0; j < 8; ++j)
            unsafeAtomicAdd(&ctxp[(size_t)(mrow + j) * DDIM + e], acc[j]);
    }
}

// ---------------------------------------------------------------- kernel 2
// q' features, d_inv = 1/(q'.kcum), out = d_inv * (q' @ ctx)
__global__ __launch_bounds__(256) void perf_qside(
    const float* __restrict__ qg, const bf16* __restrict__ projbf,
    const float* __restrict__ ctx, const float* __restrict__ kcum,
    float* __restrict__ outg) {
    extern __shared__ char smem[];
    bf16*  proj_s = (bf16*)smem;                    // [MPAD][64]
    bf16*  ctxT   = proj_s + MPAD * XPITCH;         // [64][QPITCH]
    bf16*  qp_s   = ctxT + 64 * QPITCH;             // [8][16][QPITCH]
    bf16*  xstg   = qp_s + 8 * 16 * QPITCH;         // [8][16][64]
    float* diag_s = (float*)(xstg + 8 * 16 * XPITCH);
    float* kcum_s = diag_s + 8 * 16;                // [MPAD]

    const int tid  = threadIdx.x;
    const int wave = tid >> 5;
    const int lane = tid & 31;
    const int row  = lane & 15;
    const int half = lane >> 4;
    const int bh    = blockIdx.x >> 5;
    const int chunk = blockIdx.x & (NCHUNKS - 1);

    { // proj, kcum, ctx^T -> LDS
        const uint4* src = (const uint4*)projbf;
        uint4* dst = (uint4*)proj_s;
        for (int i = tid; i < MPAD * XPITCH / 8; i += 256) dst[i] = src[i];
        for (int i = tid; i < MPAD; i += 256) kcum_s[i] = kcum[bh * MPAD + i];
        const float* ctxp = ctx + (size_t)bh * MPAD * DDIM;
        for (int i = tid; i < MPAD * DDIM; i += 256) {
            const int m = i >> 6, e = i & 63;
            ctxT[e * QPITCH + m] = (bf16)ctxp[i];
        }
    }
    __syncthreads();

    const int    n0   = chunk * CHUNK + wave * 16;
    const size_t base = (size_t)bh * NSEQ * DDIM + (size_t)n0 * DDIM;
    bf16* xs = xstg + wave * 16 * XPITCH;
    bf16* qp = qp_s + wave * 16 * QPITCH;

    stage_tile_f32(qg + base, xs, diag_s, wave, lane);
    wave_lds_fence();

    float dj[8], den[8];
#pragma unroll
    for (int j = 0; j < 8; ++j) {
        dj[j] = diag_s[wave * 16 + j + half * 8];
        den[j] = 0.f;
    }

    const v16bf a0 = load_tile(xs, XPITCH, 0,  lane);
    const v16bf a1 = load_tile(xs, XPITCH, 32, lane);

    for (int mt = 0; mt < NMT; ++mt) {
        v8f acc = {0.f, 0.f, 0.f, 0.f, 0.f, 0.f, 0.f, 0.f};
        v16bf b0 = load_tile(proj_s + mt * 16 * XPITCH, XPITCH, 0,  lane);
        v16bf b1 = load_tile(proj_s + mt * 16 * XPITCH, XPITCH, 32, lane);
        acc = wmma_bf16(a0, b0, acc);
        acc = wmma_bf16(a1, b1, acc);
        const int m  = mt * 16 + row;
        const float mRatio = (m < MFEAT) ? RATIO : 0.f;   // branchless mask
        const float kc = kcum_s[m];
#pragma unroll
        for (int j = 0; j < 8; ++j) {
            const float val = mRatio * (__expf(acc[j] - dj[j]) + KEPS);
            den[j] += val * kc;
            qp[(j + half * 8) * QPITCH + m] = (bf16)val;
        }
    }
    // reduce denominator across the 16 lanes of each half; the surviving
    // per-(lane,j) mapping equals the WMMA D-layout row, so d_inv applies
    // elementwise with no data movement.
    float dinv[8];
#pragma unroll
    for (int j = 0; j < 8; ++j) {
        float d = den[j];
        d += __shfl_xor(d, 1); d += __shfl_xor(d, 2);
        d += __shfl_xor(d, 4); d += __shfl_xor(d, 8);
        dinv[j] = 1.f / d;
    }
    wave_lds_fence();

    float* outp = outg + base;
    for (int et = 0; et < 4; ++et) {
        v8f acc = {0.f, 0.f, 0.f, 0.f, 0.f, 0.f, 0.f, 0.f};
#pragma unroll
        for (int ks = 0; ks < NMT / 2; ++ks) {       // 9 K-steps over m=288
            v16bf a = load_tile(qp, QPITCH, ks * 32, lane);
            v16bf b = load_tile(ctxT + et * 16 * QPITCH, QPITCH, ks * 32, lane);
            acc = wmma_bf16(a, b, acc);
        }
        const int e = et * 16 + row;
#pragma unroll
        for (int j = 0; j < 8; ++j)
            outp[(size_t)(j + half * 8) * DDIM + e] = acc[j] * dinv[j];
    }
}

// ---------------------------------------------------------------- launch
extern "C" void kernel_launch(void* const* d_in, const int* in_sizes, int n_in,
                              void* d_out, int out_size, void* d_ws,
                              size_t ws_size, hipStream_t stream) {
    (void)in_sizes; (void)n_in; (void)out_size; (void)ws_size;
    const float* q    = (const float*)d_in[0];
    const float* k    = (const float*)d_in[1];
    const float* v    = (const float*)d_in[2];
    const float* proj = (const float*)d_in[3];
    float* out = (float*)d_out;

    float* ctx    = (float*)d_ws;                          // [BH][MPAD][64]
    float* kcum   = ctx + (size_t)BH * MPAD * DDIM;        // [BH][MPAD]
    bf16*  projbf = (bf16*)(kcum + (size_t)BH * MPAD);     // [MPAD][64]

    const size_t smem1 =
        (size_t)(MPAD * XPITCH + 64 * KTPITCH + MPAD * KTPITCH +
                 8 * 16 * XPITCH) * 2 + (8 * 16 + MPAD) * 4;
    const size_t smem2 =
        (size_t)(MPAD * XPITCH + 64 * QPITCH + 8 * 16 * QPITCH +
                 8 * 16 * XPITCH) * 2 + (8 * 16 + MPAD) * 4;

    hipFuncSetAttribute((const void*)perf_kside,
                        hipFuncAttributeMaxDynamicSharedMemorySize, (int)smem1);
    hipFuncSetAttribute((const void*)perf_qside,
                        hipFuncAttributeMaxDynamicSharedMemorySize, (int)smem2);

    const int prep_blocks = (BH * MPAD * DDIM + 255) / 256;
    perf_prep<<<prep_blocks, 256, 0, stream>>>(proj, projbf, ctx, kcum);
    perf_kside<<<BH * NCHUNKS, 256, smem1, stream>>>(k, v, projbf, ctx, kcum);
    perf_qside<<<BH * NCHUNKS, 256, smem2, stream>>>(q, projbf, ctx, kcum, out);
}
